// MyLSTMUnit_73993696575786
// MI455X (gfx1250) — matrix operand order repaired and checked by reading, not compiled, not run on hardware
//
#include <hip/hip_runtime.h>
#include <hip/hip_bf16.h>

// ---------------------------------------------------------------------------
// LSTM cell, B=8192, E=H=1024.
// gates[B,4096] = [X|H] (bf16) @ Wc^T (bf16) + (bx+bh), then LSTM elementwise.
// fp32->bf16 conversion pass, then fused WMMA GEMM + in-wave LSTM epilogue.
// GEMM tile: block 256x128 (8 waves = 4M x 2N), wave 64M x (16h x 4 gates),
// K staged in 32-wide LDS tiles, double-buffered (1 barrier/iter).
// Staging uses GLOBAL_LOAD_ASYNC_TO_LDS_B128 (ASYNCcnt) -> zero staging VGPRs.
// ---------------------------------------------------------------------------

#define Bsz   8192
#define Esz   1024
#define Hsz   1024
#define Ksz   2048           // E + H
#define Nsz   4096           // 4 * H
#define BM    256
#define BK    32
#define NK    (Ksz / BK)     // 64

typedef __attribute__((ext_vector_type(16))) __bf16 bf16x16;
typedef __attribute__((ext_vector_type(8)))  float  floatx8;

#if defined(__has_builtin)
#  if __has_builtin(__builtin_amdgcn_global_load_async_to_lds_b128)
#    define HAVE_ASYNC_LDS 1
#  endif
#endif
#ifndef HAVE_ASYNC_LDS
#  define HAVE_ASYNC_LDS 0
#endif

#if HAVE_ASYNC_LDS
// Builtin signature (from hipcc diagnostic): (v4i as1*, v4i as3*, Ii, Ii)
typedef int v4i __attribute__((vector_size(16)));
typedef __attribute__((address_space(1))) v4i glb_v4i;
typedef __attribute__((address_space(3))) v4i lds_v4i;
#define ASYNC_B128(srcp, dstp, off)                                         \
    __builtin_amdgcn_global_load_async_to_lds_b128(                         \
        (glb_v4i*)(void*)(srcp), (lds_v4i*)(void*)(dstp), (off), 0)
__device__ __forceinline__ void wait_async0() {
#if defined(__has_builtin) && __has_builtin(__builtin_amdgcn_s_wait_asynccnt)
    __builtin_amdgcn_s_wait_asynccnt(0);
#else
    asm volatile("s_wait_asynccnt 0" ::: "memory");
#endif
}
#endif

union FragU {
    bf16x16 bf;
    uint4   u[2];
};

__device__ __forceinline__ unsigned short f2bf(float f) {
    unsigned int u = __float_as_uint(f);
    unsigned int r = u + 0x7FFFu + ((u >> 16) & 1u);   // round-to-nearest-even
    return (unsigned short)(r >> 16);
}

__device__ __forceinline__ float sigmf(float x) {
    return 1.0f / (1.0f + __expf(-x));
}

// ---------------------------------------------------------------------------
// Kernel 1: A_bf16[8192][2048] = [word_emb | ht].  grid = 8192 x 256.
// ---------------------------------------------------------------------------
__global__ void lstm_convA(const float* __restrict__ emb,
                           const float* __restrict__ ht,
                           unsigned short* __restrict__ A) {
    int i  = blockIdx.x * 256 + threadIdx.x;
    int m  = i >> 8;
    int k4 = (i & 255) << 2;

    float4 ve = *(const float4*)(emb + (size_t)m * Esz + k4);
    float4 vh = *(const float4*)(ht  + (size_t)m * Hsz + k4);

    unsigned int e0 = (unsigned int)f2bf(ve.x) | ((unsigned int)f2bf(ve.y) << 16);
    unsigned int e1 = (unsigned int)f2bf(ve.z) | ((unsigned int)f2bf(ve.w) << 16);
    unsigned int h0 = (unsigned int)f2bf(vh.x) | ((unsigned int)f2bf(vh.y) << 16);
    unsigned int h1 = (unsigned int)f2bf(vh.z) | ((unsigned int)f2bf(vh.w) << 16);

    *(uint2*)(A + (size_t)m * Ksz + k4)       = make_uint2(e0, e1);
    *(uint2*)(A + (size_t)m * Ksz + Esz + k4) = make_uint2(h0, h1);
}

// ---------------------------------------------------------------------------
// Kernel 2: Wc_bf16[4096][2048], row n = output unit, [x(1024) | h(1024)].
// grid = 4096 x 256.
// ---------------------------------------------------------------------------
__global__ void lstm_convW(const float* __restrict__ wx0, const float* __restrict__ wx1,
                           const float* __restrict__ wx2, const float* __restrict__ wx3,
                           const float* __restrict__ wh0, const float* __restrict__ wh1,
                           const float* __restrict__ wh2, const float* __restrict__ wh3,
                           unsigned short* __restrict__ W) {
    int i  = blockIdx.x * 256 + threadIdx.x;
    int n  = i >> 8;
    int k4 = (i & 255) << 2;
    int g  = n >> 10;
    int j  = n & 1023;

    const float* wx = (g == 0) ? wx0 : (g == 1) ? wx1 : (g == 2) ? wx2 : wx3;
    const float* wh = (g == 0) ? wh0 : (g == 1) ? wh1 : (g == 2) ? wh2 : wh3;

    float4 vx = *(const float4*)(wx + (size_t)j * Esz + k4);
    float4 vh = *(const float4*)(wh + (size_t)j * Hsz + k4);

    unsigned int x0 = (unsigned int)f2bf(vx.x) | ((unsigned int)f2bf(vx.y) << 16);
    unsigned int x1 = (unsigned int)f2bf(vx.z) | ((unsigned int)f2bf(vx.w) << 16);
    unsigned int h0 = (unsigned int)f2bf(vh.x) | ((unsigned int)f2bf(vh.y) << 16);
    unsigned int h1 = (unsigned int)f2bf(vh.z) | ((unsigned int)f2bf(vh.w) << 16);

    *(uint2*)(W + (size_t)n * Ksz + k4)       = make_uint2(x0, x1);
    *(uint2*)(W + (size_t)n * Ksz + Esz + k4) = make_uint2(h0, h1);
}

// ---------------------------------------------------------------------------
// Kernel 3: bsum[4096] = bx[n] + bh[n].  grid = 16 x 256.
// ---------------------------------------------------------------------------
__global__ void lstm_bias(const float* __restrict__ bx0, const float* __restrict__ bx1,
                          const float* __restrict__ bx2, const float* __restrict__ bx3,
                          const float* __restrict__ bh0, const float* __restrict__ bh1,
                          const float* __restrict__ bh2, const float* __restrict__ bh3,
                          float* __restrict__ bsum) {
    int n = blockIdx.x * 256 + threadIdx.x;
    int g = n >> 10;
    int j = n & 1023;
    const float* bx = (g == 0) ? bx0 : (g == 1) ? bx1 : (g == 2) ? bx2 : bx3;
    const float* bh = (g == 0) ? bh0 : (g == 1) ? bh1 : (g == 2) ? bh2 : bh3;
    bsum[n] = bx[j] + bh[j];
}

// ---------------------------------------------------------------------------
// Kernel 4: fused bf16 WMMA GEMM + LSTM epilogue.
//   grid = (32 n-blocks of 32 h-cols, 32 m-blocks of 256 rows), block = 256.
//   Wave layout: 8 waves = 4(M) x 2(N).  Wave tile: 64 M (4 wmma tiles) x
//   16 h-cols x 4 gates = 16 WMMAs per K-step, 16 ds_load_b128 per K-step.
//   LDS double-buffered, row stride 80 B (64 B data + 16 B pad, bank-clean).
//   Next tile staged via async VMEM->LDS (no staging VGPRs, no spills).
// ---------------------------------------------------------------------------
__global__ void __launch_bounds__(256)
lstm_gemm(const unsigned short* __restrict__ A,     // [8192][2048] bf16
          const unsigned short* __restrict__ W,     // [4096][2048] bf16
          const float* __restrict__ bsum,           // [4096]
          const float* __restrict__ ct_in,          // [8192][1024]
          float* __restrict__ out) {                // ht_new | ct_new
    __shared__ __align__(16) unsigned short Alds[2][BM * 40];   // 2 x 20 KB
    __shared__ __align__(16) unsigned short Wlds[2][128 * 40];  // 2 x 10 KB

    const int tid  = threadIdx.x;
    const int lane = tid & 31;
    const int wave = tid >> 5;
    const int wm   = wave >> 1;          // 0..3  (M, 64 rows each)
    const int wn   = wave & 1;           // 0..1  (N / h-cols, 16 each)
    const int half = lane >> 4;          // 0/1
    const int l15  = lane & 15;

    const int h0 = blockIdx.x * 32;      // h-column base for this block
    const int m0 = blockIdx.y * BM;      // row base

    // Staging: A -> thread t owns full row t (64 B per K-tile).
    //          W -> thread t owns (row = t>>1, 32 B chunk = t&1).
    const int wrowc  = tid >> 1;
    const int wchunk = tid & 1;
    const int wgate  = wrowc >> 5;
    const int wrow_n = wgate * Hsz + h0 + (wrowc & 31);

    const unsigned short* gA = A + (size_t)(m0 + tid) * Ksz;
    const unsigned short* gW = W + (size_t)wrow_n * Ksz + wchunk * 16;

    floatx8 acc[4][4];
    const floatx8 zero = {0.f, 0.f, 0.f, 0.f, 0.f, 0.f, 0.f, 0.f};
#pragma unroll
    for (int mi = 0; mi < 4; ++mi)
#pragma unroll
        for (int g = 0; g < 4; ++g) acc[mi][g] = zero;

    // ---- staging helper: move K-tile kt into LDS buffer buf -------------
    auto stage = [&](int kt, int buf) {
        const unsigned short* sa = gA + (size_t)kt * BK;
        const unsigned short* sw = gW + (size_t)kt * BK;
        char* dA = (char*)Alds[buf] + tid * 80;
        char* dW = (char*)Wlds[buf] + wrowc * 80 + wchunk * 32;
#if HAVE_ASYNC_LDS
        ASYNC_B128(sa, dA, 0);
        ASYNC_B128(sa, dA, 16);
        ASYNC_B128(sa, dA, 32);
        ASYNC_B128(sa, dA, 48);
        ASYNC_B128(sw, dW, 0);
        ASYNC_B128(sw, dW, 16);
#else
        const uint4* s4a = (const uint4*)sa;
        const uint4* s4w = (const uint4*)sw;
        uint4 a0 = s4a[0], a1 = s4a[1], a2 = s4a[2], a3 = s4a[3];
        uint4 w0 = s4w[0], w1 = s4w[1];
        uint4* d4a = (uint4*)dA;
        uint4* d4w = (uint4*)dW;
        d4a[0] = a0; d4a[1] = a1; d4a[2] = a2; d4a[3] = a3;
        d4w[0] = w0; d4w[1] = w1;
#endif
    };

    // Prologue: tile 0 -> LDS buffer 0
    stage(0, 0);
#if HAVE_ASYNC_LDS
    wait_async0();
#endif
    __syncthreads();

#pragma unroll 1
    for (int kt = 0; kt < NK; ++kt) {
        const int cur = kt & 1;
        const int nxt = cur ^ 1;

        if (kt + 1 < NK)                  // stage next tile (async, no VGPRs)
            stage(kt + 1, nxt);

        // Fragments (A and B both contiguous k = 16*half + s) + 16 WMMAs.
        FragU bf[4];
#pragma unroll
        for (int g = 0; g < 4; ++g) {
            int c = g * 32 + wn * 16 + l15;
            const char* base = (const char*)Wlds[cur] + c * 80 + half * 32;
            bf[g].u[0] = *(const uint4*)(base);          // K 0-15 / 16-31
            bf[g].u[1] = *(const uint4*)(base + 16);
        }
#pragma unroll
        for (int mi = 0; mi < 4; ++mi) {
            FragU af;
            int mrow = wm * 64 + mi * 16 + l15;
            const char* base = (const char*)Alds[cur] + mrow * 80;
            af.u[0] = *(const uint4*)(base + half * 32);       // K 0-7 / 16-23
            af.u[1] = *(const uint4*)(base + 16 + half * 32);  // K 8-15 / 24-31
#pragma unroll
            for (int g = 0; g < 4; ++g)
                acc[mi][g] = __builtin_amdgcn_wmma_f32_16x16x32_bf16(
                    false, af.bf, false, bf[g].bf,
                    (short)0, acc[mi][g], false, false);
        }

#if HAVE_ASYNC_LDS
        if (kt + 1 < NK) wait_async0();   // async writes to LDS complete
#endif
        __syncthreads();                  // publish buf[nxt]; compute(kt) done
    }

    // ------------------- LSTM epilogue (fully in-wave) ---------------------
    const int hcol = h0 + wn * 16 + l15;
    const float bf_ = bsum[0 * Hsz + hcol];
    const float bi_ = bsum[1 * Hsz + hcol];
    const float bg_ = bsum[2 * Hsz + hcol];
    const float bo_ = bsum[3 * Hsz + hcol];

#pragma unroll
    for (int mi = 0; mi < 4; ++mi) {
#pragma unroll
        for (int r = 0; r < 8; ++r) {
            int m = m0 + wm * 64 + mi * 16 + half * 8 + r;   // C-layout M index
            float fg = sigmf(acc[mi][0][r] + bf_);
            float ig = sigmf(acc[mi][1][r] + bi_);
            float gg = tanhf(acc[mi][2][r] + bg_);
            float og = sigmf(acc[mi][3][r] + bo_);
            float c_old = ct_in[(size_t)m * Hsz + hcol];
            float c_new = fg * c_old + ig * gg;
            float h_new = og * tanhf(c_new);
            out[(size_t)m * Hsz + hcol]                     = h_new;  // ht_new
            out[(size_t)Bsz * Hsz + (size_t)m * Hsz + hcol] = c_new;  // ct_new
        }
    }
}

// ---------------------------------------------------------------------------
extern "C" void kernel_launch(void* const* d_in, const int* in_sizes, int n_in,
                              void* d_out, int out_size, void* d_ws, size_t ws_size,
                              hipStream_t stream) {
    const float* word_emb = (const float*)d_in[0];
    const float* ht       = (const float*)d_in[1];
    const float* ct       = (const float*)d_in[2];
    const float* wfx_w = (const float*)d_in[3];  const float* wfx_b = (const float*)d_in[4];
    const float* wix_w = (const float*)d_in[5];  const float* wix_b = (const float*)d_in[6];
    const float* wgx_w = (const float*)d_in[7];  const float* wgx_b = (const float*)d_in[8];
    const float* wox_w = (const float*)d_in[9];  const float* wox_b = (const float*)d_in[10];
    const float* wfh_w = (const float*)d_in[11]; const float* wfh_b = (const float*)d_in[12];
    const float* wih_w = (const float*)d_in[13]; const float* wih_b = (const float*)d_in[14];
    const float* wgh_w = (const float*)d_in[15]; const float* wgh_b = (const float*)d_in[16];
    const float* woh_w = (const float*)d_in[17]; const float* woh_b = (const float*)d_in[18];

    unsigned short* Acvt = (unsigned short*)d_ws;                       // 33,554,432 B
    unsigned short* Wc   = (unsigned short*)((char*)d_ws + 33554432);   // 16,777,216 B
    float*          bsum = (float*)((char*)d_ws + 50331648);            //     16,384 B

    lstm_convA<<<dim3(8192), dim3(256), 0, stream>>>(word_emb, ht, Acvt);
    lstm_convW<<<dim3(4096), dim3(256), 0, stream>>>(wfx_w, wix_w, wgx_w, wox_w,
                                                     wfh_w, wih_w, wgh_w, woh_w, Wc);
    lstm_bias<<<dim3(16), dim3(256), 0, stream>>>(wfx_b, wix_b, wgx_b, wox_b,
                                                  wfh_b, wih_b, wgh_b, woh_b, bsum);
    lstm_gemm<<<dim3(Nsz / 128, Bsz / BM), dim3(256), 0, stream>>>(
        Acvt, Wc, bsum, ct, (float*)d_out);
}
